// MultiHeadedAttention_relative_3058016715286
// MI455X (gfx1250) — matrix operand order repaired and checked by reading, not compiled, not run on hardware
//
#include <hip/hip_runtime.h>
#include <hip/hip_bf16.h>

// ---- problem constants (reference: B,L,D,H,DK = 4,1024,512,8,64; MAX_REL=3)
#define BB  4
#define LL  1024
#define DD  512
#define HH  8
#define DKK 64
#define NROWS (BB * LL)          // 4096

typedef __attribute__((ext_vector_type(16))) _Float16 v16h;
typedef __attribute__((ext_vector_type(8)))  float    v8f;
typedef __attribute__((ext_vector_type(8)))  unsigned int v8u;

static __device__ inline v16h as_v16h(v8u u) {
    union { v8u u; v16h h; } x; x.u = u; return x.h;
}

// D = A(16x32 f16) * B(32x16 f16) + C(16x16 f32)
static __device__ inline v8f wmma16(v16h a, v16h b, v8f c) {
    return __builtin_amdgcn_wmma_f32_16x16x32_f16(
        /*neg_a=*/false, a, /*neg_b=*/false, b,
        /*c_mod=*/(short)0, c, /*reuse_a=*/false, /*reuse_b=*/false);
}

// ---- A-fragment loaders (16x32, M x K). Lane l: M = l&15; VGPR v holds K pair
//      kb = (v&3)*2 + (l>>4)*8 + (v>>2)*16  (per ISA 16-bit A layout)
static __device__ inline v16h load_a_f16(const _Float16* src, int row0, int k0,
                                         int ld, int lane) {
    const int m = lane & 15, hi = lane >> 4;
    const unsigned int* p =
        (const unsigned int*)(src + (size_t)(row0 + m) * ld + k0);
    v8u u;
#pragma unroll
    for (int v = 0; v < 4; ++v) u[v] = p[hi * 4 + v];          // K = 0..7 / 8..15
#pragma unroll
    for (int v = 0; v < 4; ++v) u[4 + v] = p[8 + hi * 4 + v];  // K = 16..23 / 24..31
    return as_v16h(u);
}

static __device__ inline v16h load_a_f32(const float* src, int row0, int k0,
                                         int ld, int lane) {
    const int m = lane & 15, hi = lane >> 4;
    const float* p = src + (size_t)(row0 + m) * ld + k0;
    v16h h;
#pragma unroll
    for (int v = 0; v < 8; ++v) {
        const int kb = (v & 3) * 2 + hi * 8 + (v >> 2) * 16;
        h[2 * v]     = (_Float16)p[kb];
        h[2 * v + 1] = (_Float16)p[kb + 1];
    }
    return h;
}

// ---- B-fragment loaders (32x16, K x N) from row-major source whose ROW n is
//      COLUMN n of B (K contiguous). Lane l: N = l&15; K = (v&3)*2 + (l>>4)*8 + (v>>2)*16 + {0,1}
static __device__ inline v16h load_b_f16(const _Float16* src, int n0, int k0,
                                         int ld, int lane) {
    const int n = lane & 15, kh = lane >> 4;
    const unsigned int* p =
        (const unsigned int*)(src + (size_t)(n0 + n) * ld + k0);
    v8u u;
#pragma unroll
    for (int v = 0; v < 8; ++v) u[v] = p[(v >> 2) * 8 + kh * 4 + (v & 3)];
    return as_v16h(u);
}

static __device__ inline v16h load_b_f32(const float* src, int n0, int k0,
                                         int ld, int lane) {
    const int n = lane & 15, kh = lane >> 4;
    const float* p = src + (size_t)(n0 + n) * ld + k0;
    v16h h;
#pragma unroll
    for (int v = 0; v < 8; ++v) {
        h[2 * v]     = (_Float16)p[(v & 3) * 2 + kh * 8 + (v >> 2) * 16];
        h[2 * v + 1] = (_Float16)p[(v & 3) * 2 + kh * 8 + (v >> 2) * 16 + 1];
    }
    return h;
}

// =====================================================================
// Projection: Y[n][d] = sum_k X[n][k] * W[d][k] + bias[d]   (torch Linear)
// Output stored f16, head-split:
//   transposed==0: out[((b*H+h)*L + l)*DK + dk]   (Q, K)
//   transposed==1: out[((b*H+h)*DK + dk)*L + l]   (V, transposed for P*V)
// One wave per 16x64 output tile (4 WMMA accumulators; A reused 4x).
// 2048 tiles / 4 waves -> 512 blocks.
// =====================================================================
__global__ __launch_bounds__(128) void proj_kernel(
    const float* __restrict__ x, const float* __restrict__ w,
    const float* __restrict__ bias, _Float16* __restrict__ out,
    int transposed) {
    const int lane = threadIdx.x & 31, wave = threadIdx.x >> 5;
    const int tile = blockIdx.x * 4 + wave;        // 0..2047
    const int rt = tile >> 3, cb = tile & 7;       // 256 row-tiles x 8 col-blocks
    const int row0 = rt << 4, col0 = cb << 6;      // 16 rows x 64 cols

    const float* arow = x + (size_t)(row0 + (lane & 15)) * DD;
    const float* brow = w + (size_t)(col0 + (lane & 15)) * DD;

    v8f acc[4];
#pragma unroll
    for (int nt = 0; nt < 4; ++nt) acc[nt] = (v8f){};

    for (int k0 = 0; k0 < DD; k0 += 32) {
        if (k0 + 32 < DD) {                         // gfx1250 global_prefetch_b8
            __builtin_prefetch(arow + k0 + 32, 0, 3);
            __builtin_prefetch(brow + k0 + 32, 0, 3);
        }
        v16h a = load_a_f32(x, row0, k0, DD, lane);
#pragma unroll
        for (int nt = 0; nt < 4; ++nt) {
            v16h b = load_b_f32(w, col0 + nt * 16, k0, DD, lane);
            acc[nt] = wmma16(a, b, acc[nt]);
        }
    }

    const int n = lane & 15, hi = lane >> 4;
#pragma unroll
    for (int nt = 0; nt < 4; ++nt) {
        const int d  = col0 + nt * 16 + n;
        const float bb = bias[d];
        const int h  = d >> 6, dk = d & (DKK - 1);
#pragma unroll
        for (int r = 0; r < 8; ++r) {
            const int row = row0 + r + 8 * hi;
            const int bi  = row >> 10, l = row & (LL - 1);
            const size_t idx = transposed
                ? ((size_t)(bi * HH + h) * DKK + dk) * LL + l
                : ((size_t)(bi * HH + h) * LL + l) * DKK + dk;
            out[idx] = (_Float16)(acc[nt][r] + bb);
        }
    }
}

// =====================================================================
// Attention: one block per (b, h, 16-row l-tile). 128 threads (4 waves).
// Dynamic LDS: scores[16][1024] f32 + out[16][64] + small scratch.
// =====================================================================
__global__ __launch_bounds__(128) void attn_kernel(
    const _Float16* __restrict__ q16, const _Float16* __restrict__ k16,
    const _Float16* __restrict__ vT16, const float* __restrict__ table_k,
    const float* __restrict__ table_v, const unsigned char* __restrict__ mask,
    _Float16* __restrict__ ao16) {
    extern __shared__ float smem[];
    float* sc   = smem;                 // 16*1024
    float* outt = sc + 16 * 1024;       // 16*64
    float* rel  = outt + 16 * 64;       // 16*8
    float* cj   = rel + 16 * 8;         // 16*8
    float* red  = cj + 16 * 8;          // 128
    float* rmax = red + 128;            // 16
    float* rsum = rmax + 16;            // 16

    const int tid = threadIdx.x, lane = tid & 31, wave = tid >> 5;
    const int bh = blockIdx.x >> 6;     // b*H + h
    const int lt = blockIdx.x & 63;
    const int b = bh >> 3, h = bh & 7;
    const int l0 = lt << 4;

    const _Float16* qbase = q16 + ((size_t)bh * LL + l0) * DKK;
    const _Float16* kbase = k16 + (size_t)bh * LL * DKK;
    const _Float16* vbase = vT16 + (size_t)bh * DKK * LL;

    // ---- scores = Q * K^T (raw dot; scale applied later with rel term)
    v16h aq0 = load_a_f16(qbase, 0, 0, DKK, lane);
    v16h aq1 = load_a_f16(qbase, 0, 32, DKK, lane);
    for (int mt = wave; mt < 64; mt += 4) {
        const int m0 = mt << 4;
        v16h bk0 = load_b_f16(kbase, m0, 0, DKK, lane);
        v16h bk1 = load_b_f16(kbase, m0, 32, DKK, lane);
        v8f c = {};
        c = wmma16(aq0, bk0, c);
        c = wmma16(aq1, bk1, c);
        const int n = lane & 15, hi = lane >> 4;
#pragma unroll
        for (int r = 0; r < 8; ++r) sc[(r + 8 * hi) * 1024 + m0 + n] = c[r];
    }
    for (int i = tid; i < 16 * 64; i += 128) outt[i] = 0.f;
    if (tid < 16 * 8) cj[tid] = 0.f;
    __syncthreads();

    // ---- rel[i][j] = q_row_i . table_k[j]   (Shaw relative-K, 7 clipped bins)
    if (tid < 112) {
        const int i = tid / 7, j = tid % 7;
        const _Float16* qr = qbase + (size_t)i * DKK;
        float acc = 0.f;
        for (int d = 0; d < DKK; ++d) acc += (float)qr[d] * table_k[j * DKK + d];
        rel[i * 8 + j] = acc;
    }
    __syncthreads();

    // ---- scores = (qk + rel)*scale, masked
    const float scale = 0.125f;  // 1/sqrt(64)
    for (int idx = tid; idx < 16 * 1024; idx += 128) {
        const int i = idx >> 10, m = idx & 1023;
        const int l = l0 + i;
        const int dlt = m - l;
        const int j = (dlt < -3 ? -3 : (dlt > 3 ? 3 : dlt)) + 3;
        float s = (sc[idx] + rel[i * 8 + j]) * scale;
        if (!mask[(size_t)(b * LL + l) * LL + m]) s = -1e9f;
        sc[idx] = s;
    }
    __syncthreads();

    // ---- softmax per row; 8 threads/row, 128-wide chunks
    {
        const int row = tid >> 3, t8 = tid & 7;
        float* srow = sc + row * 1024;
        const int base = t8 * 128;
        float pm = -3.0e38f;
        for (int mm = 0; mm < 128; ++mm) pm = fmaxf(pm, srow[base + mm]);
        red[tid] = pm;
        __syncthreads();
        if (t8 == 0) {
            float v = red[tid];
            for (int q = 1; q < 8; ++q) v = fmaxf(v, red[tid + q]);
            rmax[row] = v;
        }
        __syncthreads();
        const float rmx = rmax[row];
        float ps = 0.f;
        for (int mm = 0; mm < 128; ++mm) {
            const float e = __expf(srow[base + mm] - rmx);
            srow[base + mm] = e;
            ps += e;
        }
        red[tid] = ps;
        __syncthreads();
        if (t8 == 0) {
            float v = 0.f;
            for (int q = 0; q < 8; ++q) v += red[tid + q];
            rsum[row] = v;
        }
        __syncthreads();
        const float inv = 1.0f / rsum[row];
        const int l = l0 + row;
        float cl[7] = {0.f, 0.f, 0.f, 0.f, 0.f, 0.f, 0.f};
        for (int mm = 0; mm < 128; ++mm) {
            const float p = srow[base + mm] * inv;
            srow[base + mm] = p;
            const int dlt = base + mm - l;
            const int j = (dlt < -3 ? -3 : (dlt > 3 ? 3 : dlt)) + 3;
            cl[j] += p;    // relative-V coefficients
        }
#pragma unroll
        for (int j = 0; j < 7; ++j) atomicAdd(&cj[row * 8 + j], cl[j]);
    }
    __syncthreads();

    // ---- out = P * V via WMMA, split-K over 4 waves, ds-atomic reduce
    {
        v8f acc[4];
#pragma unroll
        for (int nt = 0; nt < 4; ++nt) acc[nt] = (v8f){};
        for (int kt = wave; kt < 32; kt += 4) {
            const int k0 = kt * 32;
            v16h pa = load_a_f32(sc, 0, k0, 1024, lane);   // P from LDS, f32->f16
#pragma unroll
            for (int nt = 0; nt < 4; ++nt) {
                v16h bvf = load_b_f16(vbase, nt * 16, k0, LL, lane);
                acc[nt] = wmma16(pa, bvf, acc[nt]);
            }
        }
        const int n = lane & 15, hi = lane >> 4;
#pragma unroll
        for (int nt = 0; nt < 4; ++nt)
#pragma unroll
            for (int r = 0; r < 8; ++r)
                atomicAdd(&outt[(r + 8 * hi) * 64 + nt * 16 + n], acc[nt][r]);
    }
    __syncthreads();

    // ---- add rank-7 relative-V correction, store f16 in [B, L, H*DK] layout
    for (int idx = tid; idx < 16 * 64; idx += 128) {
        const int i = idx >> 6, dk = idx & 63;
        float o = outt[idx];
#pragma unroll
        for (int j = 0; j < 7; ++j) o += cj[i * 8 + j] * table_v[j * DKK + dk];
        ao16[(size_t)(b * LL + l0 + i) * DD + h * DKK + dk] = (_Float16)o;
    }
}

// =====================================================================
// Output projection: out[n][d] = sum_k AO[n][k] * Wo[d][k] + bo[d], f32 store
// 16x64 tile per wave, A fragment reused across 4 B fragments.
// =====================================================================
__global__ __launch_bounds__(128) void gemm_out_kernel(
    const _Float16* __restrict__ a16, const float* __restrict__ wo,
    const float* __restrict__ bo, float* __restrict__ out) {
    const int lane = threadIdx.x & 31, wave = threadIdx.x >> 5;
    const int tile = blockIdx.x * 4 + wave;        // 0..2047
    const int rt = tile >> 3, cb = tile & 7;
    const int row0 = rt << 4, col0 = cb << 6;

    const _Float16* arow = a16 + (size_t)(row0 + (lane & 15)) * DD;
    const float*    brow = wo + (size_t)(col0 + (lane & 15)) * DD;

    v8f acc[4];
#pragma unroll
    for (int nt = 0; nt < 4; ++nt) acc[nt] = (v8f){};

    for (int k0 = 0; k0 < DD; k0 += 32) {
        if (k0 + 32 < DD) {
            __builtin_prefetch(arow + k0 + 32, 0, 3);
            __builtin_prefetch(brow + k0 + 32, 0, 3);
        }
        v16h a = load_a_f16(a16, row0, k0, DD, lane);
#pragma unroll
        for (int nt = 0; nt < 4; ++nt) {
            v16h b = load_b_f32(wo, col0 + nt * 16, k0, DD, lane);
            acc[nt] = wmma16(a, b, acc[nt]);
        }
    }

    const int n = lane & 15, hi = lane >> 4;
#pragma unroll
    for (int nt = 0; nt < 4; ++nt) {
        const float bb = bo[col0 + nt * 16 + n];
#pragma unroll
        for (int r = 0; r < 8; ++r) {
            const int row = row0 + r + 8 * hi;
            out[(size_t)row * DD + col0 + nt * 16 + n] = acc[nt][r] + bb;
        }
    }
}

// =====================================================================
extern "C" void kernel_launch(void* const* d_in, const int* in_sizes, int n_in,
                              void* d_out, int out_size, void* d_ws, size_t ws_size,
                              hipStream_t stream) {
    // setup_inputs() order:
    const float* x_q = (const float*)d_in[0];
    const float* x_k = (const float*)d_in[1];
    const float* x_v = (const float*)d_in[2];
    const float* wq  = (const float*)d_in[3];
    const float* bq  = (const float*)d_in[4];
    const float* wk  = (const float*)d_in[5];
    const float* bk  = (const float*)d_in[6];
    const float* wv  = (const float*)d_in[7];
    const float* bv  = (const float*)d_in[8];
    const float* wo  = (const float*)d_in[9];
    const float* bo  = (const float*)d_in[10];
    const float* table_k = (const float*)d_in[11];
    const float* table_v = (const float*)d_in[12];
    // d_in[13] = dist_idx (int) -- recomputed analytically on device, unused
    const unsigned char* mask = (const unsigned char*)d_in[14];

    // workspace: 4 x (4096*512) f16 buffers = 16 MB
    _Float16* q16  = (_Float16*)d_ws;
    _Float16* k16  = q16  + (size_t)NROWS * DD;
    _Float16* vT16 = k16  + (size_t)NROWS * DD;
    _Float16* ao16 = vT16 + (size_t)NROWS * DD;

    const dim3 blk(128);
    const dim3 grd(512);    // 2048 16x64 tiles / 4 waves per block

    proj_kernel<<<grd, blk, 0, stream>>>(x_q, wq, bq, q16, 0);
    proj_kernel<<<grd, blk, 0, stream>>>(x_k, wk, bk, k16, 0);
    proj_kernel<<<grd, blk, 0, stream>>>(x_v, wv, bv, vT16, 1);

    const size_t shbytes =
        (16 * 1024 + 16 * 64 + 16 * 8 + 16 * 8 + 128 + 16 + 16) * sizeof(float);
    attn_kernel<<<dim3(BB * HH * (LL / 16)), blk, shbytes, stream>>>(
        q16, k16, vT16, table_k, table_v, mask, ao16);

    gemm_out_kernel<<<grd, blk, 0, stream>>>(ao16, wo, bo, (float*)d_out);
}
